// NSF_AR_3358664426274
// MI455X (gfx1250) — matrix operand order repaired and checked by reading, not compiled, not run on hardware
//
#include <hip/hip_runtime.h>
#include <hip/hip_bf16.h>

// ---- problem constants (from reference) ----
#define NSF_K     8
#define NSF_B     3.0f
#define NSF_DIM   32
#define NSF_HID   32
#define NSF_BATCH 65536
#define NSF_L     31          // DIM-1 layers
#define NSF_P     23          // 3K-1 params per dim

typedef __attribute__((ext_vector_type(16))) _Float16 v16h;
typedef __attribute__((ext_vector_type(8)))  float    v8f;

static __device__ __forceinline__ v8f wmma_f16(v16h a, v16h b, v8f c) {
    // D = A(16x32,f16) x B(32x16,f16) + C(16x16,f32)
    return __builtin_amdgcn_wmma_f32_16x16x32_f16(
        /*neg_a=*/false, a, /*neg_b=*/false, b,
        /*c_mod=*/(short)0, c, /*reuse_a=*/false, /*reuse_b=*/false);
}

#define LOG2E_F 1.4426950408889634f
#define LN2_F   0.6931471805599453f

static __device__ __forceinline__ float fast_tanh(float x) {
    // tanh(x) = 1 - 2/(exp(2x)+1)  via v_exp_f32 (base-2) + v_rcp_f32
    float a = __builtin_amdgcn_exp2f(x * (2.0f * LOG2E_F));
    return 1.0f - 2.0f * __builtin_amdgcn_rcpf(a + 1.0f);
}

static __device__ __forceinline__ float softplusf(float x) {
    float t = __builtin_amdgcn_exp2f(x * LOG2E_F);
    float r = LN2_F * __builtin_amdgcn_logf(1.0f + t);
    return (x > 15.0f) ? x : r;
}

// double-softmax + cumwidth construction, matching the reference:
//   s  = softmax(p[0:8]); Wu = 2B*s; w = softmax(Wu);
//   widths = MIN + (1-8*MIN)*w; cw = 2B*cumsum - B; pin ends; widths = diff(cw)
static __device__ __forceinline__ void softmax_cum(const float* p, float* cb, float* wd) {
    float m = p[0];
#pragma unroll
    for (int j = 1; j < 8; ++j) m = fmaxf(m, p[j]);
    float e[8], s = 0.0f;
#pragma unroll
    for (int j = 0; j < 8; ++j) { e[j] = __builtin_amdgcn_exp2f((p[j] - m) * LOG2E_F); s += e[j]; }
    float inv = 1.0f / s;
    float q[8], m2 = -1e30f;
#pragma unroll
    for (int j = 0; j < 8; ++j) { q[j] = 6.0f * e[j] * inv; m2 = fmaxf(m2, q[j]); }
    float f[8], s2 = 0.0f;
#pragma unroll
    for (int j = 0; j < 8; ++j) { f[j] = __builtin_amdgcn_exp2f((q[j] - m2) * LOG2E_F); s2 += f[j]; }
    float inv2 = 1.0f / s2;
    float c = 0.0f;
    cb[0] = -NSF_B;
#pragma unroll
    for (int j = 0; j < 8; ++j) {
        c += 1e-3f + (1.0f - 8e-3f) * f[j] * inv2;
        cb[j + 1] = 6.0f * c - NSF_B;
    }
    cb[8] = NSF_B;
#pragma unroll
    for (int j = 0; j < 8; ++j) wd[j] = cb[j + 1] - cb[j];
}

// rational-quadratic spline forward for one (row, dim); returns z, sets logdet
static __device__ __forceinline__ float rqs_spline(const float p[NSF_P], float xin, float& ldret) {
    float cw[9], wd[8], ch[9], ht[8], dv[9];
    softmax_cum(p + 0, cw, wd);
    softmax_cum(p + 8, ch, ht);
    dv[0] = 1.0f;   // MIN_D + softplus(const) == 1 by construction (pinned edges)
    dv[8] = 1.0f;
#pragma unroll
    for (int j = 0; j < 7; ++j) dv[j + 1] = 1e-3f + softplusf(softplusf(p[16 + j]));

    bool inside = (xin >= -NSF_B) && (xin <= NSF_B);
    float xc = fminf(fmaxf(xin, -NSF_B), NSF_B);

    int idx = -1;
#pragma unroll
    for (int j = 0; j < 9; ++j) {
        float edge = (j == 8) ? (cw[8] + 1e-6f) : cw[j];
        idx += (xc >= edge) ? 1 : 0;
    }
    idx = idx < 0 ? 0 : (idx > 7 ? 7 : idx);

    float incw = cw[0], inw = wd[0], inch = ch[0], inh = ht[0], d0 = dv[0], d1 = dv[1];
#pragma unroll
    for (int j = 1; j < 8; ++j) {
        bool c = (idx == j);
        incw = c ? cw[j] : incw;  inw = c ? wd[j] : inw;
        inch = c ? ch[j] : inch;  inh = c ? ht[j] : inh;
        d0 = c ? dv[j] : d0;      d1 = c ? dv[j + 1] : d1;
    }
    float th    = (xc - incw) / inw;
    float tt    = th * (1.0f - th);
    float delta = inh / inw;
    float num   = inh * (delta * th * th + d0 * tt);
    float den   = delta + (d0 + d1 - 2.0f * delta) * tt;
    float z     = inch + num / den;
    float omt   = 1.0f - th;
    float dnum  = delta * delta * (d1 * th * th + 2.0f * delta * tt + d0 * omt * omt);
    float ldv   = LN2_F * (__builtin_amdgcn_logf(dnum) - 2.0f * __builtin_amdgcn_logf(den));
    ldret = inside ? ldv : 0.0f;
    return inside ? z : xin;
}

// ---------------------------------------------------------------------------
// One block = 8 waves x 32 batch rows = 256 rows.  Per layer & wave:
//   12x v_wmma_f32_16x16x32_f16 (full K in one WMMA), tanh, LDS-staged
//   fragment transposes, then 32 lane-parallel splines.
// ---------------------------------------------------------------------------
__global__ __launch_bounds__(256) void nsf_ar_wmma_kernel(
    const float* __restrict__ xg,  const float* __restrict__ initp,
    const float* __restrict__ w1g, const float* __restrict__ b1g,
    const float* __restrict__ w2g, const float* __restrict__ b2g,
    const float* __restrict__ w3g, const float* __restrict__ b3g,
    float* __restrict__ zout, float* __restrict__ ldout)
{
    __shared__ v16h fragBuf[6][32];                         // 6 KB  : weight B-fragments
    __shared__ __align__(16) _Float16 hstage[8][32][32];    // 16 KB : per-wave h transpose
    __shared__ float ostage[8][32][36];                     // 36 KB : per-wave spline params

    const int tid  = threadIdx.x;
    const int w    = tid >> 5;          // wave id (wave32)
    const int lane = tid & 31;
    const int half = lane >> 4;         // lane group 0/1
    const int l16  = lane & 15;
    const int g    = half * 8;          // A-fragment K interleave (ISA 16-bit A layout)
    const int rowBase = blockIdx.x * 256 + w * 32;
    const int myrow   = rowBase + lane;

    // ---- dim 0: spline with broadcast init_param ----
    float p[NSF_P];
#pragma unroll
    for (int j = 0; j < NSF_P; ++j) p[j] = initp[j];
    float ld;
    {
        float x0 = xg[myrow * NSF_DIM + 0];
        float z0 = rqs_spline(p, x0, ld);
        zout[myrow * NSF_DIM + 0] = z0;
    }

    // ---- build x A-fragments (16-bit A layout: M = lane%16, k = {g..g+7, 16+g..16+g+7}) ----
    v16h ax[2];
#pragma unroll
    for (int t = 0; t < 2; ++t) {
        const float* xr = xg + (size_t)(rowBase + t * 16 + l16) * NSF_DIM;
        float4 q0 = *(const float4*)(xr + g);
        float4 q1 = *(const float4*)(xr + g + 4);
        float4 q2 = *(const float4*)(xr + 16 + g);
        float4 q3 = *(const float4*)(xr + 20 + g);
        v16h a;
        a[0]=(_Float16)q0.x; a[1]=(_Float16)q0.y; a[2]=(_Float16)q0.z; a[3]=(_Float16)q0.w;
        a[4]=(_Float16)q1.x; a[5]=(_Float16)q1.y; a[6]=(_Float16)q1.z; a[7]=(_Float16)q1.w;
        a[8]=(_Float16)q2.x; a[9]=(_Float16)q2.y; a[10]=(_Float16)q2.z; a[11]=(_Float16)q2.w;
        a[12]=(_Float16)q3.x; a[13]=(_Float16)q3.y; a[14]=(_Float16)q3.z; a[15]=(_Float16)q3.w;
        ax[t] = a;
    }

    for (int l = 0; l < NSF_L; ++l) {
        // ---- waves 0..5 build the six weight B-fragments for this layer ----
        // B layout: lane holds n = nt*16 + lane%16, element j -> k = 8*(lane/16) + j (j<8), 16 + 8*(lane/16) + j-8 (j>=8)
        if (w < 6) {
            const int mat = w >> 1, nt = w & 1;
            const int n = nt * 16 + l16;
            v16h bf;
#pragma unroll
            for (int j = 0; j < 16; ++j) {
                const int k = (j < 8) ? (half * 8 + j) : (16 + half * 8 + (j - 8));
                float v;
                if (mat == 0)      v = (k < 31) ? w1g[((size_t)l * 31 + k) * 32 + n] : 0.0f;   // pad K row 31
                else if (mat == 1) v = w2g[((size_t)l * 32 + k) * 32 + n];
                else               v = (n < NSF_P) ? w3g[((size_t)l * 32 + k) * NSF_P + n] : 0.0f; // pad N cols
                bf[j] = (_Float16)v;
            }
            fragBuf[w][lane] = bf;
            if (l + 1 < NSF_L) {  // warm L2/L1 for next layer's weights
                __builtin_prefetch(w1g + (size_t)(l + 1) * 31 * 32, 0, 1);
                __builtin_prefetch(w2g + (size_t)(l + 1) * 32 * 32, 0, 1);
                __builtin_prefetch(w3g + (size_t)(l + 1) * 32 * NSF_P, 0, 1);
            }
        }
        __syncthreads();
        v16h bw1[2], bw2[2], bw3[2];
        bw1[0] = fragBuf[0][lane]; bw1[1] = fragBuf[1][lane];
        bw2[0] = fragBuf[2][lane]; bw2[1] = fragBuf[3][lane];
        bw3[0] = fragBuf[4][lane]; bw3[1] = fragBuf[5][lane];
        __syncthreads();   // fragments consumed -> safe to rebuild next iteration

        // ---- layer 1: h1 = tanh(x[:, :31] @ w1 + b1) ----
#pragma unroll
        for (int t = 0; t < 2; ++t) {
#pragma unroll
            for (int nt = 0; nt < 2; ++nt) {
                v8f c = {};
                c = wmma_f16(ax[t], bw1[nt], c);
                const int n = nt * 16 + l16;
                float bias = b1g[l * 32 + n];
#pragma unroll
                for (int r = 0; r < 8; ++r)
                    hstage[w][t * 16 + half * 8 + r][n] = (_Float16)fast_tanh(c[r] + bias);
            }
        }
        asm volatile("s_wait_dscnt 0" ::: "memory");   // wave-private LDS RAW (cross-lane)

        v16h ah[2];
#pragma unroll
        for (int t = 0; t < 2; ++t) {
            union { v16h v; uint4 q[2]; } u;
            u.q[0] = *(const uint4*)&hstage[w][t * 16 + l16][g];
            u.q[1] = *(const uint4*)&hstage[w][t * 16 + l16][16 + g];
            ah[t] = u.v;
        }

        // ---- layer 2: h2 = tanh(h1 @ w2 + b2) ----
#pragma unroll
        for (int t = 0; t < 2; ++t) {
#pragma unroll
            for (int nt = 0; nt < 2; ++nt) {
                v8f c = {};
                c = wmma_f16(ah[t], bw2[nt], c);
                const int n = nt * 16 + l16;
                float bias = b2g[l * 32 + n];
#pragma unroll
                for (int r = 0; r < 8; ++r)
                    hstage[w][t * 16 + half * 8 + r][n] = (_Float16)fast_tanh(c[r] + bias);
            }
        }
        asm volatile("s_wait_dscnt 0" ::: "memory");

        v16h ah2[2];
#pragma unroll
        for (int t = 0; t < 2; ++t) {
            union { v16h v; uint4 q[2]; } u;
            u.q[0] = *(const uint4*)&hstage[w][t * 16 + l16][g];
            u.q[1] = *(const uint4*)&hstage[w][t * 16 + l16][16 + g];
            ah2[t] = u.v;
        }

        // ---- layer 3: out = h2 @ w3 + b3  (params for dim l+1) ----
#pragma unroll
        for (int t = 0; t < 2; ++t) {
#pragma unroll
            for (int nt = 0; nt < 2; ++nt) {
                v8f c = {};
                c = wmma_f16(ah2[t], bw3[nt], c);
                const int n = nt * 16 + l16;
                float bias = (n < NSF_P) ? b3g[l * NSF_P + n] : 0.0f;
#pragma unroll
                for (int r = 0; r < 8; ++r)
                    ostage[w][t * 16 + half * 8 + r][n] = c[r] + bias;
            }
        }
        asm volatile("s_wait_dscnt 0" ::: "memory");

        // ---- spline for dim d = l+1, one row per lane ----
#pragma unroll
        for (int j = 0; j < NSF_P; ++j) p[j] = ostage[w][lane][j];
        float xd = xg[myrow * NSF_DIM + (l + 1)];
        float ldc;
        float z = rqs_spline(p, xd, ldc);
        zout[myrow * NSF_DIM + (l + 1)] = z;
        ld += ldc;
    }

    ldout[myrow] = ld;
}

extern "C" void kernel_launch(void* const* d_in, const int* in_sizes, int n_in,
                              void* d_out, int out_size, void* d_ws, size_t ws_size,
                              hipStream_t stream) {
    (void)in_sizes; (void)n_in; (void)out_size; (void)d_ws; (void)ws_size;
    const float* x     = (const float*)d_in[0];
    const float* initp = (const float*)d_in[1];
    const float* w1    = (const float*)d_in[2];
    const float* b1    = (const float*)d_in[3];
    const float* w2    = (const float*)d_in[4];
    const float* b2    = (const float*)d_in[5];
    const float* w3    = (const float*)d_in[6];
    const float* b3    = (const float*)d_in[7];
    float* z  = (float*)d_out;                       // [BATCH, DIM]
    float* ld = z + (size_t)NSF_BATCH * NSF_DIM;     // [BATCH]
    dim3 grid(NSF_BATCH / 256);                      // 256 rows per block (8 waves x 32)
    nsf_ar_wmma_kernel<<<grid, 256, 0, stream>>>(x, initp, w1, b1, w2, b2, w3, b3, z, ld);
}
